// GroupedQueryAttention_27264452395679
// MI455X (gfx1250) — matrix operand order repaired and checked by reading, not compiled, not run on hardware
//
#include <hip/hip_runtime.h>
#include <hip/hip_bf16.h>

// ---------------- problem constants ----------------
#define D_MODEL   2048
#define N_HEADS   16
#define N_KV      4
#define D_KH      128                 // head dim
#define KV_REP    4
#define BATCH     4
#define SEQ       2048
#define ROWS      (BATCH * SEQ)       // 8192
#define KV_DIM    (N_KV * D_KH)       // 512

typedef __attribute__((ext_vector_type(16))) __bf16 v16bf;
typedef __attribute__((ext_vector_type(8)))  float  v8f;

union FragB16 { v16bf v; unsigned short u[16]; };

__device__ __forceinline__ unsigned short f2bf(float f) {
    union { float f; unsigned u; } c; c.f = f;
    unsigned u = c.u;
    unsigned r = u + 0x7FFFu + ((u >> 16) & 1u);   // round-to-nearest-even
    return (unsigned short)(r >> 16);
}
__device__ __forceinline__ float bf2f(unsigned short h) {
    union { unsigned u; float f; } c; c.u = ((unsigned)h) << 16;
    return c.f;
}

// ---- CDNA5 async global->LDS copy (ASYNCcnt path, no VGPR round-trip) ----
// GVS mode: mem = SADDR(64b sgpr pair) + VADDR(32b byte offset); VDST = LDS byte addr.
__device__ __forceinline__ void async_copy_b128(const void* gbase, unsigned goff,
                                                void* lptr) {
    unsigned lds = (unsigned)(unsigned long long)lptr;  // generic low 32 bits == LDS offset
    asm volatile("global_load_async_to_lds_b128 %0, %1, %2 offset:0"
                 :: "v"(lds), "v"(goff), "s"((unsigned long long)gbase)
                 : "memory");
}
__device__ __forceinline__ void wait_async_all() {
    asm volatile("s_wait_asynccnt 0x0" ::: "memory");
}

// ---------------- fp32 -> bf16 convert ----------------
__global__ void cvt_f32_to_bf16(const float* __restrict__ src,
                                unsigned short* __restrict__ dst, int n) {
    int i = blockIdx.x * blockDim.x + threadIdx.x;
    if (i < n) dst[i] = f2bf(src[i]);
}

// ---------------- tiled bf16 WMMA GEMM: C = A(MxK) @ B(KxN) + bias ----------------
// block = 256 threads (8 waves); 128x128 tile; each wave: 64x32 (4x2 subtiles).
// Double-buffered async LDS staging.
template <bool WRITE_F32>
__global__ __launch_bounds__(256)
void gemm_bf16_wmma(const unsigned short* __restrict__ A,
                    const unsigned short* __restrict__ Bw,
                    const float* __restrict__ bias,
                    unsigned short* __restrict__ Cb,
                    float* __restrict__ Cf,
                    int M, int N, int K) {
    (void)M;
    __shared__ __align__(16) unsigned short Xs[2][128 * 32];
    __shared__ __align__(16) unsigned short Ws[2][32 * 128];

    const int t    = threadIdx.x;
    const int lane = t & 31;
    const int wave = t >> 5;
    const int wy   = wave >> 2;      // 0..1
    const int wx   = wave & 3;       // 0..3
    const int hl   = lane >> 4;      // half-wave selector
    const int lm   = lane & 15;

    const int rowBase = blockIdx.y * 128;
    const int colBase = blockIdx.x * 128;

    const v8f vzero = {};
    v8f acc[4][2];
#pragma unroll
    for (int m = 0; m < 4; ++m)
#pragma unroll
        for (int n = 0; n < 2; ++n) acc[m][n] = vzero;

    const int xr = t >> 1;           // 0..127
    const int xc = (t & 1) << 4;     // 0/16
    const int wr = t >> 3;           // 0..31
    const int wc = (t & 7) << 4;     // 0..112

    const unsigned xbase = (unsigned)(((unsigned)(rowBase + xr) * (unsigned)K + xc) * 2u);
    const unsigned wbase = (unsigned)(((unsigned)wr * (unsigned)N + colBase + wc) * 2u);

    auto stage = [&](int p, int k0) {
        unsigned xo = xbase + (unsigned)k0 * 2u;
        async_copy_b128(A, xo,      &Xs[p][xr * 32 + xc]);
        async_copy_b128(A, xo + 16, &Xs[p][xr * 32 + xc + 8]);
        unsigned wo = wbase + (unsigned)k0 * (unsigned)N * 2u;
        async_copy_b128(Bw, wo,      &Ws[p][wr * 128 + wc]);
        async_copy_b128(Bw, wo + 16, &Ws[p][wr * 128 + wc + 8]);
    };

    stage(0, 0);

    int p = 0;
    for (int k0 = 0; k0 < K; k0 += 32, p ^= 1) {
        wait_async_all();            // this wave's copies for buffer p have landed
        __syncthreads();             // everyone's copies visible; prev reads of p^1 done
        if (k0 + 32 < K) stage(p ^ 1, k0 + 32);

        FragB16 af[4], bf[2];
#pragma unroll
        for (int m = 0; m < 4; ++m)
#pragma unroll
            for (int i = 0; i < 16; ++i) {
                int kk = (hl << 4) + i;                              // A layout
                af[m].u[i] = Xs[p][(wy * 64 + m * 16 + lm) * 32 + kk];
            }
#pragma unroll
        for (int n = 0; n < 2; ++n)
#pragma unroll
            for (int i = 0; i < 16; ++i) {
                int kk = (hl << 4) + i;                              // B layout
                bf[n].u[i] = Ws[p][kk * 128 + wx * 32 + n * 16 + lm];
            }
#pragma unroll
        for (int m = 0; m < 4; ++m)
#pragma unroll
            for (int n = 0; n < 2; ++n)
                acc[m][n] = __builtin_amdgcn_wmma_f32_16x16x32_bf16(
                    false, af[m].v, false, bf[n].v, (short)0, acc[m][n], false, false);
    }

#pragma unroll
    for (int m = 0; m < 4; ++m)
#pragma unroll
        for (int n = 0; n < 2; ++n) {
            int col = colBase + wx * 32 + n * 16 + lm;
            float bv = bias[col];
#pragma unroll
            for (int r = 0; r < 8; ++r) {
                int row = rowBase + wy * 64 + m * 16 + r + hl * 8;
                float val = acc[m][n][r] + bv;
                if (WRITE_F32) Cf[(size_t)row * N + col] = val;
                else           Cb[(size_t)row * N + col] = f2bf(val);
            }
        }
}

// ---------------- in-place RoPE on bf16 buffer (rows x heads*128) ----------------
__global__ void rope_inplace(unsigned short* __restrict__ buf, int heads, int total) {
    int id = blockIdx.x * blockDim.x + threadIdx.x;
    if (id >= total) return;
    int d   = id & 63;
    int hh  = (id >> 6) % heads;
    int row = id / (heads * 64);
    int tp  = row % SEQ;
    size_t base = (size_t)row * (heads * D_KH) + hh * D_KH;
    float u1 = bf2f(buf[base + d]);
    float u2 = bf2f(buf[base + d + 64]);
    float inv = __powf(10000.0f, -(float)d * (1.0f / 64.0f));
    float ang = (float)tp * inv;
    float cs = __cosf(ang), sn = __sinf(ang);
    buf[base + d]      = f2bf(u1 * cs - u2 * sn);
    buf[base + d + 64] = f2bf(u2 * cs + u1 * sn);
}

// ---------------- flash attention (causal, GQA) ----------------
// grid: (SEQ/128, BATCH*N_HEADS); block 256 (8 waves).
// wave w owns query rows qBase + w*16 .. +15; streams 32-key blocks via
// double-buffered async LDS staging.
__global__ __launch_bounds__(256)
void attn_flash(const unsigned short* __restrict__ Qb,
                const unsigned short* __restrict__ Kb,
                const unsigned short* __restrict__ Vb,
                unsigned short* __restrict__ Ob) {
    __shared__ __align__(16) unsigned short Ks[2][32 * 128];
    __shared__ __align__(16) unsigned short Vs[2][32 * 128];
    __shared__ unsigned short Ps[8][16 * 32];

    const int t    = threadIdx.x;
    const int lane = t & 31;
    const int wave = t >> 5;
    const int hl   = lane >> 4;
    const int lm   = lane & 15;

    const int bh  = blockIdx.y;
    const int b   = bh / N_HEADS;
    const int h   = bh % N_HEADS;
    const int kvh = h / KV_REP;
    const int qBase = blockIdx.x * 128;
    const int qRow0 = qBase + wave * 16;
    const float scale = 0.08838834764831845f;      // 1/sqrt(128)

    // Q fragments for this wave's 16 rows (16x128 -> 4 A-fragments)
    FragB16 qf[4];
    {
        size_t qoff = (size_t)(b * SEQ + qRow0 + lm) * D_MODEL + h * D_KH;
#pragma unroll
        for (int c = 0; c < 4; ++c)
#pragma unroll
            for (int i = 0; i < 16; ++i) {
                int kk = (hl << 4) + i;
                qf[c].u[i] = Qb[qoff + c * 32 + kk];
            }
    }

    const v8f vzero = {};
    v8f o[8];
#pragma unroll
    for (int n3 = 0; n3 < 8; ++n3) o[n3] = vzero;
    float mrow[8], lrow[8];
#pragma unroll
    for (int r = 0; r < 8; ++r) { mrow[r] = -1e30f; lrow[r] = 0.0f; }

    const int kr = t >> 3;           // 0..31
    const int kc = (t & 7) << 4;     // 0..112
    const unsigned kvbase =
        (unsigned)((((unsigned)(b * SEQ + kr)) * KV_DIM + kvh * D_KH + kc) * 2u);
    const int nBlocks = qBase / 32 + 4;

    auto stageKV = [&](int p, int j) {
        unsigned go = kvbase + (unsigned)j * 32u * KV_DIM * 2u;
        async_copy_b128(Kb, go,      &Ks[p][kr * 128 + kc]);
        async_copy_b128(Kb, go + 16, &Ks[p][kr * 128 + kc + 8]);
        async_copy_b128(Vb, go,      &Vs[p][kr * 128 + kc]);
        async_copy_b128(Vb, go + 16, &Vs[p][kr * 128 + kc + 8]);
    };

    stageKV(0, 0);

    int p = 0;
    for (int j = 0; j < nBlocks; ++j, p ^= 1) {
        wait_async_all();
        __syncthreads();
        if (j + 1 < nBlocks) stageKV(p ^ 1, j + 1);

        // S = Q @ K^T : two 16x16 subtiles (32 keys)
        v8f s0v = vzero, s1v = vzero;
#pragma unroll
        for (int c = 0; c < 4; ++c) {
            FragB16 kf;
#pragma unroll
            for (int i = 0; i < 16; ++i) {
                int kk = (hl << 4) + i;
                kf.u[i] = Ks[p][(0 * 16 + lm) * 128 + c * 32 + kk];
            }
            s0v = __builtin_amdgcn_wmma_f32_16x16x32_bf16(
                false, qf[c].v, false, kf.v, (short)0, s0v, false, false);
        }
#pragma unroll
        for (int c = 0; c < 4; ++c) {
            FragB16 kf;
#pragma unroll
            for (int i = 0; i < 16; ++i) {
                int kk = (hl << 4) + i;
                kf.u[i] = Ks[p][(1 * 16 + lm) * 128 + c * 32 + kk];
            }
            s1v = __builtin_amdgcn_wmma_f32_16x16x32_bf16(
                false, qf[c].v, false, kf.v, (short)0, s1v, false, false);
        }

        // online softmax per owned row (fully unrolled: no dynamic VGPR indexing)
#pragma unroll
        for (int r = 0; r < 8; ++r) {
            int q   = qRow0 + r + hl * 8;
            int k0i = j * 32 + lm;
            int k1i = j * 32 + 16 + lm;
            float s0 = s0v[r] * scale;
            float s1 = s1v[r] * scale;
            bool m0 = k0i > q, m1 = k1i > q;
            float mx = fmaxf(m0 ? -1e30f : s0, m1 ? -1e30f : s1);
            mx = fmaxf(mx, __shfl_xor(mx, 1, 32));
            mx = fmaxf(mx, __shfl_xor(mx, 2, 32));
            mx = fmaxf(mx, __shfl_xor(mx, 4, 32));
            mx = fmaxf(mx, __shfl_xor(mx, 8, 32));
            float mnew = fmaxf(mrow[r], mx);
            float p0 = m0 ? 0.0f : __expf(s0 - mnew);
            float p1 = m1 ? 0.0f : __expf(s1 - mnew);
            float rs = p0 + p1;
            rs += __shfl_xor(rs, 1, 32);
            rs += __shfl_xor(rs, 2, 32);
            rs += __shfl_xor(rs, 4, 32);
            rs += __shfl_xor(rs, 8, 32);
            float alpha = __expf(mrow[r] - mnew);
            lrow[r] = lrow[r] * alpha + rs;
            mrow[r] = mnew;
#pragma unroll
            for (int n3 = 0; n3 < 8; ++n3) o[n3][r] = o[n3][r] * alpha;
            int rowId = r + hl * 8;
            Ps[wave][rowId * 32 + lm]      = f2bf(p0);
            Ps[wave][rowId * 32 + 16 + lm] = f2bf(p1);
        }

        // transpose P through per-wave LDS into an A-fragment (16x32)
        FragB16 pf;
#pragma unroll
        for (int i = 0; i < 16; ++i) {
            int kk = (hl << 4) + i;
            pf.u[i] = Ps[wave][lm * 32 + kk];
        }
        // O += P @ V (32-key slab, 8 dim-subtiles)
#pragma unroll
        for (int n3 = 0; n3 < 8; ++n3) {
            FragB16 vf;
#pragma unroll
            for (int i = 0; i < 16; ++i) {
                int kk = (hl << 4) + i;
                vf.u[i] = Vs[p][kk * 128 + n3 * 16 + lm];
            }
            o[n3] = __builtin_amdgcn_wmma_f32_16x16x32_bf16(
                false, pf.v, false, vf.v, (short)0, o[n3], false, false);
        }
    }

#pragma unroll
    for (int n3 = 0; n3 < 8; ++n3)
#pragma unroll
        for (int r = 0; r < 8; ++r) {
            int q = qRow0 + r + hl * 8;
            float val = o[n3][r] / lrow[r];
            Ob[(size_t)(b * SEQ + q) * D_MODEL + h * D_KH + n3 * 16 + lm] = f2bf(val);
        }
}

// ---------------- host-side orchestration ----------------
extern "C" void kernel_launch(void* const* d_in, const int* in_sizes, int n_in,
                              void* d_out, int out_size, void* d_ws, size_t ws_size,
                              hipStream_t stream) {
    (void)in_sizes; (void)n_in; (void)out_size; (void)ws_size;
    const float* x    = (const float*)d_in[0];
    const float* wq_w = (const float*)d_in[1];
    const float* wq_b = (const float*)d_in[2];
    const float* wk_w = (const float*)d_in[3];
    const float* wk_b = (const float*)d_in[4];
    const float* wv_w = (const float*)d_in[5];
    const float* wv_b = (const float*)d_in[6];
    const float* o_w  = (const float*)d_in[7];
    const float* o_b  = (const float*)d_in[8];
    float* out = (float*)d_out;

    char* ws = (char*)d_ws;
    size_t off = 0;
    auto alloc = [&](size_t bytes) {
        void* p = ws + off;
        off += (bytes + 255) & ~(size_t)255;
        return p;
    };
    unsigned short* xb  = (unsigned short*)alloc((size_t)ROWS * D_MODEL * 2);
    unsigned short* wqb = (unsigned short*)alloc((size_t)D_MODEL * D_MODEL * 2);
    unsigned short* wkb = (unsigned short*)alloc((size_t)D_MODEL * KV_DIM * 2);
    unsigned short* wvb = (unsigned short*)alloc((size_t)D_MODEL * KV_DIM * 2);
    unsigned short* owb = (unsigned short*)alloc((size_t)D_MODEL * D_MODEL * 2);
    unsigned short* Qb  = (unsigned short*)alloc((size_t)ROWS * D_MODEL * 2);
    unsigned short* Kb  = (unsigned short*)alloc((size_t)ROWS * KV_DIM * 2);
    unsigned short* Vb  = (unsigned short*)alloc((size_t)ROWS * KV_DIM * 2);
    unsigned short* Ab  = (unsigned short*)alloc((size_t)ROWS * D_MODEL * 2);

    auto cvt = [&](const float* s, unsigned short* d, size_t n) {
        int nn = (int)n;
        cvt_f32_to_bf16<<<dim3((nn + 255) / 256), dim3(256), 0, stream>>>(s, d, nn);
    };
    cvt(x,    xb,  (size_t)ROWS * D_MODEL);
    cvt(wq_w, wqb, (size_t)D_MODEL * D_MODEL);
    cvt(wk_w, wkb, (size_t)D_MODEL * KV_DIM);
    cvt(wv_w, wvb, (size_t)D_MODEL * KV_DIM);
    cvt(o_w,  owb, (size_t)D_MODEL * D_MODEL);

    gemm_bf16_wmma<false><<<dim3(D_MODEL / 128, ROWS / 128), 256, 0, stream>>>(
        xb, wqb, wq_b, Qb, nullptr, ROWS, D_MODEL, D_MODEL);
    gemm_bf16_wmma<false><<<dim3(KV_DIM / 128, ROWS / 128), 256, 0, stream>>>(
        xb, wkb, wk_b, Kb, nullptr, ROWS, KV_DIM, D_MODEL);
    gemm_bf16_wmma<false><<<dim3(KV_DIM / 128, ROWS / 128), 256, 0, stream>>>(
        xb, wvb, wv_b, Vb, nullptr, ROWS, KV_DIM, D_MODEL);

    rope_inplace<<<dim3((ROWS * N_HEADS * 64 + 255) / 256), 256, 0, stream>>>(
        Qb, N_HEADS, ROWS * N_HEADS * 64);
    rope_inplace<<<dim3((ROWS * N_KV * 64 + 255) / 256), 256, 0, stream>>>(
        Kb, N_KV, ROWS * N_KV * 64);

    attn_flash<<<dim3(SEQ / 128, BATCH * N_HEADS), 256, 0, stream>>>(Qb, Kb, Vb, Ab);

    gemm_bf16_wmma<true><<<dim3(D_MODEL / 128, ROWS / 128), 256, 0, stream>>>(
        Ab, owb, o_b, nullptr, out, ROWS, D_MODEL, D_MODEL);
}
